// FlashBaichuanAttention_15942918602783
// MI455X (gfx1250) — matrix operand order repaired and checked by reading, not compiled, not run on hardware
//
#include <hip/hip_runtime.h>
#include <cstdint>

// ---------------------------------------------------------------------------
// MI455X (gfx1250) fused attention: QKV GEMM -> RoPE -> causal flash attn ->
// output GEMM.  All matmuls use v_wmma_f32_16x16x32_bf16 (bf16 in, f32 acc).
// Global->LDS staging uses GLOBAL_LOAD_ASYNC_TO_LDS_B128 (ASYNCcnt) — the
// builtin exists in this toolchain (probe: expects v4i __device__* / v4i
// __shared__* args), with a VGPR-bounce fallback otherwise.
// B=2, S=2048, HID=4096, H=32, D=128.
// ---------------------------------------------------------------------------

typedef __attribute__((ext_vector_type(16))) __bf16 v16bf;
typedef __attribute__((ext_vector_type(8)))  float  v8f;
typedef int v4i_vec __attribute__((vector_size(16)));   // matches builtin param

union FragBF {
    v16bf v;
    uint4 u[2];
};

#if defined(__AMDGCN__) && __has_builtin(__builtin_amdgcn_global_load_async_to_lds_b128)
#define HAVE_ASYNC_LDS 1
#endif

// one 16-byte global -> LDS copy (async when available)
__device__ __forceinline__ void async_copy_b128(unsigned short* ldsDst,
                                                const unsigned short* gSrc) {
#ifdef HAVE_ASYNC_LDS
    __builtin_amdgcn_global_load_async_to_lds_b128(
        (__attribute__((address_space(1))) v4i_vec*)gSrc,
        (__attribute__((address_space(3))) v4i_vec*)ldsDst,
        /*offset=*/0, /*cpol=*/0);
#else
    *(uint4*)ldsDst = *(const uint4*)gSrc;
#endif
}

__device__ __forceinline__ void async_copies_wait() {
#ifdef HAVE_ASYNC_LDS
#if __has_builtin(__builtin_amdgcn_s_wait_asynccnt)
    __builtin_amdgcn_s_wait_asynccnt(0);
#else
    asm volatile("s_wait_asynccnt 0x0" ::: "memory");
#endif
#endif
}

__device__ __forceinline__ unsigned short f2bf(float f) {
    unsigned int u = __float_as_uint(f);
    u += 0x7FFFu + ((u >> 16) & 1u);   // round-to-nearest-even
    return (unsigned short)(u >> 16);
}
__device__ __forceinline__ float bf2f(unsigned short h) {
    return __uint_as_float(((unsigned int)h) << 16);
}

// ---- fragment loaders --------------------------------------------------
// A fragment (16x32, row-major source, ld = elements/row).
// ISA 16-bit A layout: lanes 0-15 hold row M=lane, K in {0..7,16..23};
// lanes 16-31 hold row M=lane-16, K in {8..15,24..31}. Two b128 loads/lane.
__device__ __forceinline__ v16bf frag_ld_A(const unsigned short* p, int ld) {
    const int lane = threadIdx.x & 31;
    const int r = lane & 15, hi = lane >> 4;
    const unsigned short* q = p + (size_t)r * ld + hi * 8;
    FragBF f;
    f.u[0] = *(const uint4*)(q);
    f.u[1] = *(const uint4*)(q + 16);
    return f.v;
}
// B fragment (32x16): source is a [16 x 32] row-major tile whose row n is
// column n of B (i.e. weights stored [N][K], K contiguous).  Per sparse-B
// ISA layout: lanes 0-15 hold K in {0..7,16..23}, lanes 16-31 hold K in {8..15,24..31}.
__device__ __forceinline__ v16bf frag_ld_B(const unsigned short* p, int ld) {
    const int lane = threadIdx.x & 31;
    const int n = lane & 15, hi = lane >> 4;
    const unsigned short* q = p + (size_t)n * ld + hi * 8;
    FragBF f;
    f.u[0] = *(const uint4*)(q);
    f.u[1] = *(const uint4*)(q + 16);
    return f.v;
}

__device__ __forceinline__ v8f wmma_bf16(v16bf a, v16bf b, v8f c) {
    return __builtin_amdgcn_wmma_f32_16x16x32_bf16(
        /*neg_a=*/false, a, /*neg_b=*/false, b,
        /*c_mod=*/(short)0, c, /*reuse_a=*/false, /*reuse_b=*/false);
}

// ---------------------------------------------------------------------------
// fp32 -> bf16 conversion (n divisible by 256)
// ---------------------------------------------------------------------------
__global__ void cvt_f32_bf16_kernel(const float* __restrict__ in,
                                    unsigned short* __restrict__ out) {
    size_t i = (size_t)blockIdx.x * 256 + threadIdx.x;
    out[i] = f2bf(in[i]);
}

// ---------------------------------------------------------------------------
// GEMM: C[M,N] = A[M,K] * W[N,K]^T   (A,W bf16; C bf16 or f32)
// 128x128 block tile, BK=64, 256 threads = 8 waves in a 2x4 grid,
// each wave computes a 32x64 sub-tile (2x4 WMMA accumulators).
// Double-buffered LDS with async global->LDS copies: stage tile k+1 while
// WMMAs consume tile k; one s_wait_asynccnt per iteration.
// ---------------------------------------------------------------------------
__global__ __launch_bounds__(256) void gemm_bf16_kernel(
    const unsigned short* __restrict__ A,
    const unsigned short* __restrict__ W,
    void* __restrict__ Cout,
    int M, int N, int K, int outIsF32)
{
    __shared__ __align__(16) unsigned short lA[2][128 * 72];
    __shared__ __align__(16) unsigned short lB[2][128 * 72];

    const int tid  = threadIdx.x;
    const int lane = tid & 31;
    const int w    = tid >> 5;
    const int wr   = w >> 1;       // 0..3  (M direction, 32 rows each)
    const int wc   = w & 1;        // 0..1  (N direction, 64 cols each)
    const int hi   = lane >> 4;
    const size_t bm = (size_t)blockIdx.y * 128;
    const size_t bn = (size_t)blockIdx.x * 128;

    auto stage = [&](int buf, int k0) {
#pragma unroll
        for (int i = 0; i < 4; ++i) {
            int idx = tid + i * 256;          // 0..1023
            int row = idx >> 3;               // 0..127
            int cv  = (idx & 7) * 8;          // 0..56
            async_copy_b128(&lA[buf][row * 72 + cv],
                            &A[(bm + row) * K + k0 + cv]);
            async_copy_b128(&lB[buf][row * 72 + cv],
                            &W[(bn + row) * K + k0 + cv]);
        }
    };

    v8f acc[2][4];
#pragma unroll
    for (int mt = 0; mt < 2; ++mt)
#pragma unroll
        for (int nt = 0; nt < 4; ++nt)
            acc[mt][nt] = v8f{};

    stage(0, 0);                              // preload first tile
    int buf = 0;
    for (int k0 = 0; k0 < K; k0 += 64, buf ^= 1) {
        async_copies_wait();                  // this wave's copies landed
        __syncthreads();                      // everyone's copies visible
        if (k0 + 64 < K)
            stage(buf ^ 1, k0 + 64);          // overlap next tile with compute

#pragma unroll
        for (int ks = 0; ks < 64; ks += 32) {
            v16bf af[2], bfr[4];
#pragma unroll
            for (int mt = 0; mt < 2; ++mt)
                af[mt] = frag_ld_A(&lA[buf][(wr * 32 + mt * 16) * 72 + ks], 72);
#pragma unroll
            for (int nt = 0; nt < 4; ++nt)
                bfr[nt] = frag_ld_B(&lB[buf][(wc * 64 + nt * 16) * 72 + ks], 72);
#pragma unroll
            for (int mt = 0; mt < 2; ++mt)
#pragma unroll
                for (int nt = 0; nt < 4; ++nt)
                    acc[mt][nt] = wmma_bf16(af[mt], bfr[nt], acc[mt][nt]);
        }
    }

    // C layout: element i of lane L is C[mtile*16 + 8*hi + i][ntile*16 + (L&15)]
#pragma unroll
    for (int mt = 0; mt < 2; ++mt)
#pragma unroll
        for (int nt = 0; nt < 4; ++nt)
#pragma unroll
            for (int i = 0; i < 8; ++i) {
                size_t row = bm + wr * 32 + mt * 16 + hi * 8 + i;
                size_t col = bn + wc * 64 + nt * 16 + (lane & 15);
                float vv = acc[mt][nt][i];
                if (outIsF32)
                    ((float*)Cout)[row * N + col] = vv;
                else
                    ((unsigned short*)Cout)[row * N + col] = f2bf(vv);
            }
}

// ---------------------------------------------------------------------------
// RoPE in place on q,k halves of qkv[4096][12288] (bf16).
// one thread per (m, h, d2); handles both q and k.
// ---------------------------------------------------------------------------
__global__ void rope_kernel(unsigned short* __restrict__ qkv) {
    unsigned int id = blockIdx.x * 256u + threadIdx.x;   // < 4096*32*64
    int d2 = id & 63;
    int h  = (id >> 6) & 31;
    int m  = id >> 11;
    int s  = m & 2047;
    // inv_freq = 10000^(-d2/64) = exp(-ln(1e4) * d2/64)
    float freq = __expf(-(float)d2 * (9.210340371976184f / 64.0f));
    float ang = (float)s * freq;
    float sn, cs;
    __sincosf(ang, &sn, &cs);

    unsigned short* base = qkv + (size_t)m * 12288 + h * 128;
#pragma unroll
    for (int qk = 0; qk < 2; ++qk) {
        float x1 = bf2f(base[d2]);
        float x2 = bf2f(base[64 + d2]);
        base[d2]      = f2bf(x1 * cs - x2 * sn);
        base[64 + d2] = f2bf(x2 * cs + x1 * sn);
        base += 4096;
    }
}

// ---------------------------------------------------------------------------
// Causal flash attention. Block = 4 waves (128 thr) = 64 query rows of one
// (b,h). Each wave owns 16 rows. Key/value blocks of 32 staged in LDS:
// K rows async-copied, V transposed through VGPRs (so P*V B-fragments are
// contiguous b128 loads).
// ---------------------------------------------------------------------------
__global__ __launch_bounds__(128) void flash_attn_kernel(
    const unsigned short* __restrict__ qkv,
    unsigned short* __restrict__ attn)
{
    const int S = 2048, HID = 4096, E = 12288, D = 128;
    const int bh = blockIdx.y;
    const int b  = bh >> 5;
    const int h  = bh & 31;
    const int q0blk = blockIdx.x * 64;

    const int tid  = threadIdx.x;
    const int lane = tid & 31;
    const int w    = tid >> 5;
    const int r    = lane & 15;
    const int hi   = lane >> 4;
    const int q0   = q0blk + w * 16;

    __shared__ __align__(16) unsigned short lK[32 * 136];
    __shared__ __align__(16) unsigned short lVT[128 * 40];
    __shared__ __align__(16) unsigned short lP[4][16 * 32];

    const unsigned short* qbase = qkv + (size_t)(b * S) * E + h * D;
    const unsigned short* kbase = qbase + HID;
    const unsigned short* vbase = qbase + 2 * HID;

    // Q fragments (16 rows x 128 K, 4 chunks of 32), straight from global.
    v16bf qf[4];
#pragma unroll
    for (int kc = 0; kc < 4; ++kc) {
        const unsigned short* p =
            qbase + (size_t)(q0 + r) * E + kc * 32 + hi * 8;
        FragBF f;
        f.u[0] = *(const uint4*)(p);
        f.u[1] = *(const uint4*)(p + 16);
        qf[kc] = f.v;
    }

    v8f o[8];
#pragma unroll
    for (int dt = 0; dt < 8; ++dt) o[dt] = v8f{};
    float mrun[8], lrun[8];
#pragma unroll
    for (int i = 0; i < 8; ++i) { mrun[i] = -1e30f; lrun[i] = 0.0f; }

    const float scale = 0.08838834764831845f;   // 1/sqrt(128)
    const int nblk = q0blk / 32 + 2;            // keys through q0blk+63

    for (int blk = 0; blk < nblk; ++blk) {
        const int j0 = blk * 32;
        __syncthreads();
        // K block: 32 rows x 128 cols, row-major, async b128 copies.
#pragma unroll
        for (int i = 0; i < 4; ++i) {
            int idx = tid + i * 128;            // 0..511
            int kr = idx >> 4;
            int c  = (idx & 15) * 8;
            async_copy_b128(&lK[kr * 136 + c],
                            &kbase[(size_t)(j0 + kr) * E + c]);
        }
        // V block transposed: lVT[d][key], so B-frag rows are contiguous.
#pragma unroll
        for (int i = 0; i < 16; ++i) {
            int idx = tid + i * 128;            // 0..2047
            int vr = idx >> 6;                  // key 0..31
            int c2 = idx & 63;                  // dword col
            unsigned int u =
                *(const unsigned int*)&vbase[(size_t)(j0 + vr) * E + c2 * 2];
            lVT[(c2 * 2 + 0) * 40 + vr] = (unsigned short)(u & 0xFFFFu);
            lVT[(c2 * 2 + 1) * 40 + vr] = (unsigned short)(u >> 16);
        }
        async_copies_wait();                    // K copies landed in LDS
        __syncthreads();

        if (j0 <= q0 + 15) {                    // wave-uniform: EXEC stays full
            // scores: 16 q-rows x 32 keys, two 16x16 C tiles.
            v8f s0 = v8f{}, s1 = v8f{};
#pragma unroll
            for (int kc = 0; kc < 4; ++kc) {
                s0 = wmma_bf16(qf[kc], frag_ld_B(&lK[0  * 136 + kc * 32], 136), s0);
                s1 = wmma_bf16(qf[kc], frag_ld_B(&lK[16 * 136 + kc * 32], 136), s1);
            }
            const int col0 = j0 + (lane & 15);
            const int col1 = col0 + 16;
#pragma unroll
            for (int i = 0; i < 8; ++i) {
                int row = q0 + hi * 8 + i;
                float a = s0[i] * scale;
                float c = s1[i] * scale;
                if (col0 > row) a = -1e30f;     // causal mask
                if (col1 > row) c = -1e30f;
                // row max across the 16 lanes of this half-wave
                float mx = fmaxf(a, c);
#pragma unroll
                for (int d = 1; d < 16; d <<= 1)
                    mx = fmaxf(mx, __shfl_xor(mx, d, 32));
                float mnew = fmaxf(mrun[i], mx);
                float fs = __expf(mrun[i] - mnew);
                a = __expf(a - mnew);
                c = __expf(c - mnew);
                float rs = a + c;
#pragma unroll
                for (int d = 1; d < 16; d <<= 1)
                    rs += __shfl_xor(rs, d, 32);
                lrun[i] = lrun[i] * fs + rs;
                mrun[i] = mnew;
#pragma unroll
                for (int dt = 0; dt < 8; ++dt) o[dt][i] *= fs;
                // P (C-layout) -> row-major LDS scratch for A-frag re-swizzle
                unsigned short* pr = &lP[w][(hi * 8 + i) * 32];
                pr[(lane & 15)]      = f2bf(a);
                pr[16 + (lane & 15)] = f2bf(c);
            }
            // O += P(16x32) x V(32x128): 8 WMMAs over d-tiles
            v16bf pf = frag_ld_A(&lP[w][0], 32);
#pragma unroll
            for (int dt = 0; dt < 8; ++dt)
                o[dt] = wmma_bf16(pf, frag_ld_B(&lVT[(dt * 16) * 40], 40), o[dt]);
        }
    }

    // epilogue: O /= l ; write attn[b, s, h*D + d] as bf16
#pragma unroll
    for (int dt = 0; dt < 8; ++dt)
#pragma unroll
        for (int i = 0; i < 8; ++i) {
            size_t row = (size_t)b * S + q0 + hi * 8 + i;
            size_t col = (size_t)h * D + dt * 16 + (lane & 15);
            attn[row * HID + col] = f2bf(o[dt][i] / lrun[i]);
        }
}

// ---------------------------------------------------------------------------
// launch
// ---------------------------------------------------------------------------
extern "C" void kernel_launch(void* const* d_in, const int* in_sizes, int n_in,
                              void* d_out, int out_size, void* d_ws, size_t ws_size,
                              hipStream_t stream) {
    const float* x     = (const float*)d_in[0];   // [2,2048,4096]
    const float* w_qkv = (const float*)d_in[1];   // [12288,4096]
    const float* w_o   = (const float*)d_in[2];   // [4096,4096]
    float* out = (float*)d_out;                   // [2,2048,4096] fp32

    const size_t M = 4096, K = 4096, N1 = 12288, N2 = 4096;

    unsigned short* ws    = (unsigned short*)d_ws;
    unsigned short* xb    = ws;                       // 16,777,216
    unsigned short* wqkvb = xb    + M * K;            // 50,331,648
    unsigned short* wob   = wqkvb + N1 * K;           // 16,777,216
    unsigned short* qkvb  = wob   + N2 * K;           // 50,331,648
    unsigned short* attb  = qkvb  + M * N1;           // 16,777,216

    // fp32 -> bf16
    cvt_f32_bf16_kernel<<<(M * K) / 256, 256, 0, stream>>>(x, xb);
    cvt_f32_bf16_kernel<<<(N1 * K) / 256, 256, 0, stream>>>(w_qkv, wqkvb);
    cvt_f32_bf16_kernel<<<(N2 * K) / 256, 256, 0, stream>>>(w_o, wob);

    // qkv = x @ w_qkv^T   (bf16 out)
    gemm_bf16_kernel<<<dim3(N1 / 128, M / 128), 256, 0, stream>>>(
        xb, wqkvb, (void*)qkvb, (int)M, (int)N1, (int)K, 0);

    // RoPE on q,k in place
    rope_kernel<<<(4096u * 32u * 64u) / 256u, 256, 0, stream>>>(qkvb);

    // causal flash attention -> attb bf16 [4096, 4096]
    flash_attn_kernel<<<dim3(2048 / 64, 2 * 32), 128, 0, stream>>>(qkvb, attb);

    // out = attn @ w_o^T  (fp32 out)
    gemm_bf16_kernel<<<dim3(N2 / 128, M / 128), 256, 0, stream>>>(
        attb, wob, (void*)out, (int)M, (int)N2, (int)K, 1);
}
